// Autoformer_84550726189195
// MI455X (gfx1250) — compile-verified, matrix operand order checked by reading, not run on hardware
//
#include <hip/hip_runtime.h>
#include <hip/hip_bf16.h>
#include <math.h>

// ---- Problem constants (from reference) ----
#define BB    32
#define LL    1024
#define CIN   21
#define DM    512
#define DFF   2048
#define ELAY  3
#define NCLS  10
#define MV    (BB * LL)        // 32768 rows of activations
#define TOPK  6                // int(1 * log(1024)) = 6
#define CHM   8192             // FFN M-chunk (4 chunks)

typedef __attribute__((ext_vector_type(16))) __bf16 v16bf;
typedef __attribute__((ext_vector_type(8)))  float  v8f;

union Frag16 { unsigned int u[8]; v16bf v; };
union Acc8   { v8f v; float f[8]; };

__device__ __forceinline__ unsigned short f32_to_bf16(float f) {
    unsigned int u = __builtin_bit_cast(unsigned int, f);
    unsigned int r = u + 0x7FFFu + ((u >> 16) & 1u);   // RNE
    return (unsigned short)(r >> 16);
}

__device__ __forceinline__ float gelu_exact(float x) {
    return 0.5f * x * (1.0f + erff(x * 0.70710678118654752f));
}

// ---------------- fp32 -> bf16 conversion ----------------
__global__ void cvt_f32_bf16(const float* __restrict__ src,
                             unsigned short* __restrict__ dst, int n) {
    int i = blockIdx.x * blockDim.x + threadIdx.x;
    if (i < n) dst[i] = f32_to_bf16(src[i]);
}

// Load one 16x32 bf16 fragment (CDNA5 ISA 7.12.2 16-bit A layout; B mirrored
// over N): per lane two contiguous b128 loads.
__device__ __forceinline__ void load_frag(Frag16& f, const unsigned int* row, int dw) {
    *(uint4*)&f.u[0] = *(const uint4*)(row + dw);       // K {0..7}/{8..15}
    *(uint4*)&f.u[4] = *(const uint4*)(row + dw + 8);   // K {16..23}/{24..31}
}

__device__ __forceinline__ void store_tile(const Acc8& acc, int row0, int col, int hi,
                                           int N, const float* bias, const float* resid,
                                           float* outF, unsigned short* outBF, int act) {
    const float bval = bias ? bias[col] : 0.0f;
    #pragma unroll
    for (int p = 0; p < 8; ++p) {                 // C/D: VGPR p -> M = p + hi*8
        const int row = row0 + p + hi * 8;
        const size_t o = (size_t)row * N + col;
        float y = acc.f[p] + bval;
        if (resid) y += resid[o];
        if (act == 1) y = gelu_exact(y);
        if (outF)  outF[o]  = y;
        if (outBF) outBF[o] = f32_to_bf16(y);
    }
}

// ---------------- WMMA GEMM: Y = act(A[M,K] * W[N,K]^T + bias + resid) ------
// 32x32 register tile per wave: 2 A-frags x 2 B-frags -> 4 accumulators,
// 2 b128 loads per v_wmma_f32_16x16x32_bf16 (16 flop/byte from L2).
__global__ __launch_bounds__(256)
void wmma_gemm_bf16(const unsigned short* __restrict__ A,
                    const unsigned short* __restrict__ Bw,
                    const float* __restrict__ bias,
                    const float* __restrict__ resid,
                    float* __restrict__ outF,
                    unsigned short* __restrict__ outBF,
                    int M, int N, int K, int act) {
    const int lane = threadIdx.x & 31;
    const int ntn  = N >> 5;                           // 32-wide N tiles
    const int tile = blockIdx.x * 8 + threadIdx.y;     // wave-uniform
    if (tile >= (M >> 5) * ntn) return;                // whole-wave exit: EXEC all-ones
    const int m0 = (tile / ntn) << 5;
    const int n0 = (tile % ntn) << 5;
    const int r  = lane & 15;
    const int hi = lane >> 4;

    const unsigned int* a0row = (const unsigned int*)(A  + (size_t)(m0 + r)      * K);
    const unsigned int* a1row = (const unsigned int*)(A  + (size_t)(m0 + 16 + r) * K);
    const unsigned int* b0row = (const unsigned int*)(Bw + (size_t)(n0 + r)      * K);
    const unsigned int* b1row = (const unsigned int*)(Bw + (size_t)(n0 + 16 + r) * K);

    Acc8 c00, c01, c10, c11;
    #pragma unroll
    for (int p = 0; p < 8; ++p) { c00.f[p] = 0.f; c01.f[p] = 0.f; c10.f[p] = 0.f; c11.f[p] = 0.f; }

    for (int k0 = 0; k0 < K; k0 += 32) {
        const int dw = (k0 >> 1) + hi * 4;
        Frag16 fa0, fa1, fb0, fb1;
        load_frag(fa0, a0row, dw);
        load_frag(fa1, a1row, dw);
        load_frag(fb0, b0row, dw);
        load_frag(fb1, b1row, dw);
        if (k0 + 32 < K) {                   // prefetch next K-step of the A streams
            __builtin_prefetch(a0row + dw + 16, 0, 3);
            __builtin_prefetch(a1row + dw + 16, 0, 3);
        }
        c00.v = __builtin_amdgcn_wmma_f32_16x16x32_bf16(false, fa0.v, false, fb0.v, (short)0, c00.v, false, false);
        c01.v = __builtin_amdgcn_wmma_f32_16x16x32_bf16(false, fa0.v, false, fb1.v, (short)0, c01.v, false, false);
        c10.v = __builtin_amdgcn_wmma_f32_16x16x32_bf16(false, fa1.v, false, fb0.v, (short)0, c10.v, false, false);
        c11.v = __builtin_amdgcn_wmma_f32_16x16x32_bf16(false, fa1.v, false, fb1.v, (short)0, c11.v, false, false);
    }

    store_tile(c00, m0,      n0 + r,      hi, N, bias, resid, outF, outBF, act);
    store_tile(c01, m0,      n0 + 16 + r, hi, N, bias, resid, outF, outBF, act);
    store_tile(c10, m0 + 16, n0 + r,      hi, N, bias, resid, outF, outBF, act);
    store_tile(c11, m0 + 16, n0 + 16 + r, hi, N, bias, resid, outF, outBF, act);
}

// ---------------- circular Conv1d embedding (C=21 -> D=512, k=3) ------------
__global__ __launch_bounds__(128)
void conv_embed(const float* __restrict__ x, const float* __restrict__ w,
                float* __restrict__ y) {
    const int bl = blockIdx.x;              // b*L + l
    const int b = bl >> 10, l = bl & 1023;
    __shared__ float xs[63];
    const int t = threadIdx.x;
    if (t < 63) {
        const int j = t / 21, c = t % 21;
        const int ls = (l + j - 1 + LL) & (LL - 1);   // circular pad
        xs[t] = x[((size_t)b * LL + ls) * CIN + c];
    }
    __syncthreads();
    #pragma unroll
    for (int i = 0; i < 4; ++i) {
        const int dm = t + i * 128;
        const float* wr = w + (size_t)dm * 63;        // [dm][c][j]
        float s = 0.0f;
        #pragma unroll
        for (int c = 0; c < 21; ++c)
            #pragma unroll
            for (int j = 0; j < 3; ++j)
                s += xs[j * 21 + c] * wr[c * 3 + j];
        y[((size_t)b * LL + l) * DM + dm] = s;
    }
}

// -------- mean autocorrelation: mv[b,tau] = (1/512) sum_t <q[t+tau], k[t]> --
__global__ __launch_bounds__(256)
void mean_corr(const float* __restrict__ q, const float* __restrict__ k,
               float* __restrict__ mv) {
    const int b = blockIdx.x >> 10, tau = blockIdx.x & 1023;
    const float* qb = q + (size_t)b * LL * DM;
    const float* kb = k + (size_t)b * LL * DM;
    float s = 0.0f;
    for (int t = threadIdx.x; t < LL; t += 256) {
        const float4* qr = (const float4*)(qb + (size_t)((t + tau) & 1023) * DM);
        const float4* kr = (const float4*)(kb + (size_t)t * DM);
        for (int d = 0; d < DM / 4; ++d) {
            const float4 a = qr[d], c = kr[d];
            s += a.x * c.x + a.y * c.y + a.z * c.z + a.w * c.w;
        }
    }
    __shared__ float red[256];
    red[threadIdx.x] = s; __syncthreads();
    for (int st = 128; st > 0; st >>= 1) {
        if (threadIdx.x < st) red[threadIdx.x] += red[threadIdx.x + st];
        __syncthreads();
    }
    if (threadIdx.x == 0) mv[(size_t)b * LL + tau] = red[0] * (1.0f / 512.0f);
}

// ---------------- top-6 + softmax over mean correlations --------------------
__global__ __launch_bounds__(256)
void topk_softmax(const float* __restrict__ mv, float* __restrict__ topw,
                  int* __restrict__ topd) {
    const int b = blockIdx.x;
    __shared__ float vals[LL];
    __shared__ float rmax[256];
    __shared__ int   ridx[256];
    __shared__ float selw[TOPK];
    __shared__ int   seld[TOPK];
    for (int i = threadIdx.x; i < LL; i += 256) vals[i] = mv[(size_t)b * LL + i];
    __syncthreads();
    for (int it = 0; it < TOPK; ++it) {
        float m = -1e30f; int mi = 0;
        for (int i = threadIdx.x; i < LL; i += 256) {
            const float v = vals[i];
            if (v > m) { m = v; mi = i; }
        }
        rmax[threadIdx.x] = m; ridx[threadIdx.x] = mi; __syncthreads();
        for (int st = 128; st > 0; st >>= 1) {
            if (threadIdx.x < st && rmax[threadIdx.x + st] > rmax[threadIdx.x]) {
                rmax[threadIdx.x] = rmax[threadIdx.x + st];
                ridx[threadIdx.x] = ridx[threadIdx.x + st];
            }
            __syncthreads();
        }
        if (threadIdx.x == 0) {
            selw[it] = rmax[0]; seld[it] = ridx[0];
            vals[ridx[0]] = -1e30f;
        }
        __syncthreads();
    }
    if (threadIdx.x == 0) {
        const float mx = selw[0];
        float e[TOPK], sum = 0.0f;
        for (int i = 0; i < TOPK; ++i) { e[i] = __expf(selw[i] - mx); sum += e[i]; }
        for (int i = 0; i < TOPK; ++i) {
            topw[b * TOPK + i] = e[i] / sum;
            topd[b * TOPK + i] = seld[i];
        }
    }
}

// ---------------- time-delay aggregation -> bf16 ----------------------------
// Stages the 6 delayed V rows into LDS with CDNA5 async global->LDS copies
// (ASYNCcnt-tracked), then reduces from LDS.
__global__ __launch_bounds__(256)
void time_delay_agg(const float* __restrict__ v, const float* __restrict__ topw,
                    const int* __restrict__ topd, unsigned short* __restrict__ aggbf) {
    const int b = blockIdx.x >> 10, l = blockIdx.x & 1023;
    const int tid = threadIdx.x;
    __shared__ float rows[TOPK][DM];
    __shared__ float wsh[TOPK];
    if (tid < TOPK) wsh[tid] = topw[b * TOPK + tid];

    const float* vb = v + (size_t)b * LL * DM;
    #pragma unroll
    for (int i = 0; i < TOPK; ++i) {
        const int idx = (l + topd[b * TOPK + i]) & 1023;
        const float* src = vb + (size_t)idx * DM + tid * 2;     // 8B per lane
        const unsigned ldsoff =
            (unsigned)(reinterpret_cast<uintptr_t>(&rows[i][tid * 2]));
        asm volatile("global_load_async_to_lds_b64 %0, %1, off"
                     :: "v"(ldsoff), "v"(src) : "memory");
    }
    asm volatile("s_wait_asynccnt 0x0" ::: "memory");
    __syncthreads();

    for (int d = tid; d < DM; d += 256) {
        float s = 0.0f;
        #pragma unroll
        for (int i = 0; i < TOPK; ++i) s += wsh[i] * rows[i][d];
        aggbf[((size_t)b * LL + l) * DM + d] = f32_to_bf16(s);
    }
}

// ---------------- series decomposition residual: x - movavg25(x) ------------
__global__ void series_decomp_sub(const float* __restrict__ x,
                                  float* __restrict__ out, int n) {
    const int i = blockIdx.x * blockDim.x + threadIdx.x;
    if (i >= n) return;
    const int d = i & (DM - 1);
    const int bl = i >> 9;
    const int l = bl & 1023, b = bl >> 10;
    const float* xb = x + (size_t)b * LL * DM + d;
    float s = 0.0f;
    #pragma unroll
    for (int j = -12; j <= 12; ++j) {
        int lj = l + j;
        lj = lj < 0 ? 0 : (lj > 1023 ? 1023 : lj);    // replicate pad
        s += xb[(size_t)lj * DM];
    }
    out[i] = x[i] - s * (1.0f / 25.0f);
}

// ---------------- layernorm (per row) ---------------------------------------
__global__ __launch_bounds__(256)
void layernorm_row(const float* __restrict__ x, const float* __restrict__ g,
                   const float* __restrict__ bta, float* __restrict__ xh) {
    const int row = blockIdx.x;
    const float* xr = x + (size_t)row * DM;
    const float a0 = xr[threadIdx.x], a1 = xr[threadIdx.x + 256];
    __shared__ float red[256];
    red[threadIdx.x] = a0 + a1; __syncthreads();
    for (int st = 128; st > 0; st >>= 1) {
        if (threadIdx.x < st) red[threadIdx.x] += red[threadIdx.x + st];
        __syncthreads();
    }
    const float mu = red[0] * (1.0f / 512.0f);
    __syncthreads();
    const float d0 = a0 - mu, d1 = a1 - mu;
    red[threadIdx.x] = d0 * d0 + d1 * d1; __syncthreads();
    for (int st = 128; st > 0; st >>= 1) {
        if (threadIdx.x < st) red[threadIdx.x] += red[threadIdx.x + st];
        __syncthreads();
    }
    const float rstd = rsqrtf(red[0] * (1.0f / 512.0f) + 1e-5f);
    float* xo = xh + (size_t)row * DM;
    xo[threadIdx.x]       = d0 * rstd * g[threadIdx.x]       + bta[threadIdx.x];
    xo[threadIdx.x + 256] = d1 * rstd * g[threadIdx.x + 256] + bta[threadIdx.x + 256];
}

// ---------------- mean over sequence axis -----------------------------------
__global__ void seq_mean(const float* __restrict__ xh, float* __restrict__ cm) {
    const int i = blockIdx.x * blockDim.x + threadIdx.x;   // b*DM + d
    if (i >= BB * DM) return;
    const int b = i >> 9, d = i & 511;
    const float* p = xh + (size_t)b * LL * DM + d;
    float s = 0.0f;
    for (int l = 0; l < LL; ++l) s += p[(size_t)l * DM];
    cm[i] = s * (1.0f / 1024.0f);
}

// ---------------- (xh - seqmean) -> gelu -> * mark --------------------------
__global__ void final_act(const float* __restrict__ xh, const float* __restrict__ cm,
                          const float* __restrict__ mark, float* __restrict__ act, int n) {
    const int i = blockIdx.x * blockDim.x + threadIdx.x;
    if (i >= n) return;
    const int d = i & 511;
    const int bl = i >> 9, b = bl >> 10;
    const float v = xh[i] - cm[b * DM + d];
    act[i] = gelu_exact(v) * mark[bl];
}

// ---------------- classifier head -------------------------------------------
__global__ __launch_bounds__(256)
void classify(const float* __restrict__ act, const float* __restrict__ Wp,
              const float* __restrict__ bp, float* __restrict__ out) {
    const int b = blockIdx.x / NCLS, c = blockIdx.x % NCLS;
    const float4* a = (const float4*)(act + (size_t)b * LL * DM);
    const float4* w = (const float4*)(Wp + (size_t)c * LL * DM);
    float s = 0.0f;
    for (int i = threadIdx.x; i < LL * DM / 4; i += 256) {
        const float4 x = a[i], y = w[i];
        s += x.x * y.x + x.y * y.y + x.z * y.z + x.w * y.w;
    }
    __shared__ float red[256];
    red[threadIdx.x] = s; __syncthreads();
    for (int st = 128; st > 0; st >>= 1) {
        if (threadIdx.x < st) red[threadIdx.x] += red[threadIdx.x + st];
        __syncthreads();
    }
    if (threadIdx.x == 0) out[b * NCLS + c] = red[0] + bp[c];
}

// ============================================================================
extern "C" void kernel_launch(void* const* d_in, const int* in_sizes, int n_in,
                              void* d_out, int out_size, void* d_ws, size_t ws_size,
                              hipStream_t stream) {
    const float* x_enc   = (const float*)d_in[0];
    const float* x_mark  = (const float*)d_in[1];
    const float* conv_w  = (const float*)d_in[2];
    const float* Wq      = (const float*)d_in[3];
    const float* bq      = (const float*)d_in[4];
    const float* Wk      = (const float*)d_in[5];
    const float* bk      = (const float*)d_in[6];
    const float* Wv      = (const float*)d_in[7];
    const float* bv      = (const float*)d_in[8];
    const float* Wo      = (const float*)d_in[9];
    const float* bo      = (const float*)d_in[10];
    const float* Wff1    = (const float*)d_in[11];
    const float* Wff2    = (const float*)d_in[12];
    const float* ln_g    = (const float*)d_in[13];
    const float* ln_b    = (const float*)d_in[14];
    const float* Wproj   = (const float*)d_in[15];
    const float* bproj   = (const float*)d_in[16];
    float* out = (float*)d_out;

    // ---- workspace carving (aligned 256B) ----
    size_t off = 0;
    auto carve = [&](size_t bytes) -> char* {
        char* p = (char*)d_ws + off;
        off += (bytes + 255) & ~(size_t)255;
        return p;
    };
    float*          X    = (float*)carve((size_t)MV * DM * 4);          // activation
    float*          Q    = (float*)carve((size_t)MV * DM * 4);          // q / x+a / xh
    float*          Kb   = (float*)carve((size_t)MV * DM * 4);          // k / x1 / act
    float*          V    = (float*)carve((size_t)MV * DM * 4);          // v / y2
    unsigned short* XBF  = (unsigned short*)carve((size_t)MV * DM * 2); // x_bf / agg_bf
    unsigned short* HBF  = (unsigned short*)carve((size_t)CHM * DFF * 2);
    unsigned short* WQb  = (unsigned short*)carve((size_t)ELAY * DM * DM * 2);
    unsigned short* WKb  = (unsigned short*)carve((size_t)ELAY * DM * DM * 2);
    unsigned short* WVb  = (unsigned short*)carve((size_t)ELAY * DM * DM * 2);
    unsigned short* WOb  = (unsigned short*)carve((size_t)ELAY * DM * DM * 2);
    unsigned short* WF1b = (unsigned short*)carve((size_t)ELAY * DFF * DM * 2);
    unsigned short* WF2b = (unsigned short*)carve((size_t)ELAY * DM * DFF * 2);
    float*          MVal = (float*)carve((size_t)BB * LL * 4);
    float*          TW   = (float*)carve((size_t)BB * TOPK * 4);
    int*            TD   = (int*)carve((size_t)BB * TOPK * 4);
    float*          CM   = (float*)carve((size_t)BB * DM * 4);
    (void)ws_size; (void)n_in; (void)in_sizes; (void)out_size;

    auto cvt = [&](const float* s, unsigned short* d, int n) {
        cvt_f32_bf16<<<(n + 255) / 256, 256, 0, stream>>>(s, d, n);
    };
    auto gemm = [&](const unsigned short* A, const unsigned short* Bw,
                    const float* bias, const float* resid,
                    float* oF, unsigned short* oBF, int M, int N, int K, int act) {
        const int tiles = (M / 32) * (N / 32);
        wmma_gemm_bf16<<<(tiles + 7) / 8, dim3(32, 8), 0, stream>>>(
            A, Bw, bias, resid, oF, oBF, M, N, K, act);
    };

    // ---- one-time weight conversion to bf16 ----
    cvt(Wq,   WQb,  ELAY * DM * DM);
    cvt(Wk,   WKb,  ELAY * DM * DM);
    cvt(Wv,   WVb,  ELAY * DM * DM);
    cvt(Wo,   WOb,  ELAY * DM * DM);
    cvt(Wff1, WF1b, ELAY * DFF * DM);
    cvt(Wff2, WF2b, ELAY * DM * DFF);

    // ---- token embedding ----
    conv_embed<<<MV, 128, 0, stream>>>(x_enc, conv_w, X);

    // ---- encoder layers ----
    for (int l = 0; l < ELAY; ++l) {
        const unsigned short* wq = WQb  + (size_t)l * DM * DM;
        const unsigned short* wk = WKb  + (size_t)l * DM * DM;
        const unsigned short* wv = WVb  + (size_t)l * DM * DM;
        const unsigned short* wo = WOb  + (size_t)l * DM * DM;
        const unsigned short* w1 = WF1b + (size_t)l * DFF * DM;
        const unsigned short* w2 = WF2b + (size_t)l * DM * DFF;

        cvt(X, XBF, MV * DM);
        gemm(XBF, wq, bq + l * DM, nullptr, Q,  nullptr, MV, DM, DM, 0);
        gemm(XBF, wk, bk + l * DM, nullptr, Kb, nullptr, MV, DM, DM, 0);
        gemm(XBF, wv, bv + l * DM, nullptr, V,  nullptr, MV, DM, DM, 0);

        mean_corr<<<MV, 256, 0, stream>>>(Q, Kb, MVal);
        topk_softmax<<<BB, 256, 0, stream>>>(MVal, TW, TD);
        time_delay_agg<<<MV, 256, 0, stream>>>(V, TW, TD, XBF);   // agg -> bf16

        // x = x + a  (residual fused)
        gemm(XBF, wo, bo + l * DM, X, Q, nullptr, MV, DM, DM, 0);
        // x1 = decomp(x + a)
        series_decomp_sub<<<(MV * DM) / 256, 256, 0, stream>>>(Q, Kb, MV * DM);

        // FFN, chunked over M to keep hidden bf16 buffer at 32 MB
        cvt(Kb, XBF, MV * DM);
        for (int c = 0; c < MV / CHM; ++c) {
            const size_t ro = (size_t)c * CHM;
            gemm(XBF + ro * DM, w1, nullptr, nullptr, nullptr, HBF,
                 CHM, DFF, DM, /*gelu*/1);
            gemm(HBF, w2, nullptr, Kb + ro * DM, V + ro * DM, nullptr,
                 CHM, DM, DFF, 0);
        }
        // x = decomp(x1 + y)
        series_decomp_sub<<<(MV * DM) / 256, 256, 0, stream>>>(V, X, MV * DM);
    }

    // ---- head: special LN, seq-mean removal, gelu*mask, projection ----
    layernorm_row<<<MV, 256, 0, stream>>>(X, ln_g, ln_b, Q);
    seq_mean<<<(BB * DM + 255) / 256, 256, 0, stream>>>(Q, CM);
    final_act<<<(MV * DM) / 256, 256, 0, stream>>>(Q, CM, x_mark, Kb, MV * DM);
    classify<<<BB * NCLS, 256, 0, stream>>>(Kb, Wproj, bproj, out);
}
